// ClassicRNN_12206297055897
// MI455X (gfx1250) — matrix operand order repaired and checked by reading, not compiled, not run on hardware
//
#include <hip/hip_runtime.h>

// ---------------- problem constants ----------------
#define BB   256
#define TT   512
#define VV   64
#define HH   256
#define INW  129          // 2V+1
#define KTX  5            // x-GEMM k-tiles (IN padded 129 -> 160)
#define KTH  8            // h-GEMM k-tiles (H/32)
#define NT3  48           // 3H/16 column tiles
#define XS   168          // LDS stride for X (bf16, padded)
#define HS   264          // LDS stride for H (padded)
#define NTHR 512          // 16 wave32 per workgroup

typedef __attribute__((ext_vector_type(16))) __bf16 v16bf;
typedef __attribute__((ext_vector_type(8)))  float  v8f;

union FragU { uint4 u[2]; v16bf v; };

// A fragment: two 16B LDS loads 32B apart (16-bit A 16x32 layout, ISA 7.12.2)
__device__ __forceinline__ v16bf ldAf(const __bf16* p) {
    FragU f;
    f.u[0] = *(const uint4*)(p);
    f.u[1] = *(const uint4*)(p + 16);
    return f.v;
}
// B fragment: 32 contiguous bytes per lane from the packed weight stream
__device__ __forceinline__ v16bf ldBf(const __bf16* p) {
    FragU f;
    f.u[0] = *(const uint4*)(p);
    f.u[1] = *(const uint4*)(p + 8);
    return f.v;
}

#define WMMA_BF16(A, Bm, C) \
    __builtin_amdgcn_wmma_f32_16x16x32_bf16(false, (A), false, (Bm), (short)0, (C), false, false)

__device__ __forceinline__ float sigmoidf_(float x) {
    return 1.0f / (1.0f + __expf(-x));
}
__device__ __forceinline__ float tanhf_(float x) {
    return 1.0f - 2.0f / (__expf(2.0f * x) + 1.0f);
}

// ---------------- weight packing kernels ----------------
__global__ void pack_whh(const float* __restrict__ W, __bf16* __restrict__ out) {
    int idx = blockIdx.x * 256 + threadIdx.x;          // NT3*KTH*32*16 = 196608
    if (idx >= NT3 * KTH * 32 * 16) return;
    int e    = idx & 15;
    int lane = (idx >> 4) & 31;
    int kt   = (idx >> 9) & 7;
    int nt   = idx >> 12;
    int n = nt * 16 + (lane & 15);
    int k = kt * 32 + ((lane < 16) ? 0 : 8) + (e & 7) + ((e & 8) ? 16 : 0);
    out[idx] = (__bf16)W[n * HH + k];
}

__global__ void pack_wih(const float* __restrict__ W, __bf16* __restrict__ out) {
    int idx = blockIdx.x * 256 + threadIdx.x;          // NT3*KTX*32*16 = 122880
    if (idx >= NT3 * KTX * 32 * 16) return;
    int e    = idx & 15;
    int lane = (idx >> 4) & 31;
    int r    = idx >> 9;
    int kt   = r % KTX;
    int nt   = r / KTX;
    int n = nt * 16 + (lane & 15);
    int k = kt * 32 + ((lane < 16) ? 0 : 8) + (e & 7) + ((e & 8) ? 16 : 0);
    out[idx] = (k < INW) ? (__bf16)W[n * INW + k] : (__bf16)0.0f;
}

// ---------------- fused persistent GRU-D scan ----------------
// grid = 16 WGs, 512 threads (16 wave32). WG g owns batch rows [16g,16g+16).
// Wave w owns h-columns [16w, 16w+16) across all three gates.
__global__ void __launch_bounds__(NTHR, 1)
grud_scan(const float* __restrict__ times_in,
          const float* __restrict__ data_in,
          const float* __restrict__ mask_in,
          const float* __restrict__ b_ih,
          const float* __restrict__ b_hh,
          const __bf16* __restrict__ whhp,
          const __bf16* __restrict__ wihp,
          float* __restrict__ out) {
    __shared__ __bf16 Xlds[16 * XS];
    __shared__ __bf16 Hbf [16 * HS];
    __shared__ float  Hf  [16 * HS];
    __shared__ float  texist[16];

    const int tid   = threadIdx.x;
    const int wv    = tid >> 5;          // wave 0..15
    const int lane  = tid & 31;
    const int lcol  = lane & 15;
    const int bBase = blockIdx.x * 16;

    // init LDS (h0 = 0, X pad cols = 0)
    for (int i = tid; i < 16 * HS; i += NTHR) { Hbf[i] = (__bf16)0.0f; Hf[i] = 0.0f; }
    for (int i = tid; i < 16 * XS; i += NTHR) { Xlds[i] = (__bf16)0.0f; }

    // hoisted biases for column j = 16w + lcol
    const int j = 16 * wv + lcol;
    const float bR  = b_ih[j]        + b_hh[j];
    const float bZ  = b_ih[HH + j]   + b_hh[HH + j];
    const float bNX = b_ih[2*HH + j];
    const float bNH = b_hh[2*HH + j];

    // loop-invariant per-lane fragment base pointers
    const int aoff = (lane & 16) ? 8 : 0;
    const __bf16* const aXb = Xlds + (lane & 15) * XS + aoff;
    const __bf16* const aHb = Hbf  + (lane & 15) * HS + aoff;
    const __bf16* const pXR = wihp + (((size_t)( wv      * KTX) * 32 + lane) << 4);
    const __bf16* const pXZ = wihp + (((size_t)((16 + wv) * KTX) * 32 + lane) << 4);
    const __bf16* const pXN = wihp + (((size_t)((32 + wv) * KTX) * 32 + lane) << 4);
    const __bf16* const pHR = whhp + (((size_t)( wv      * KTH) * 32 + lane) << 4);
    const __bf16* const pHZ = whhp + (((size_t)((16 + wv) * KTH) * 32 + lane) << 4);
    const __bf16* const pHN = whhp + (((size_t)((32 + wv) * KTH) * 32 + lane) << 4);

    // gate-phase per-lane LDS pointers (rows rowBase..rowBase+7, column j)
    const int rowBase = (lane & 16) ? 8 : 0;
    float*  const hP  = &Hf [rowBase * HS + j];
    __bf16* const hbP = &Hbf[rowBase * HS + j];
    const float* const teP = &texist[rowBase];
    __syncthreads();

    for (int t = 0; t < TT; ++t) {
        // ---- stage x_t = [values | mask | delta] as bf16 in LDS ----
#pragma unroll
        for (int kk = 0; kk < 4; ++kk) {
            const int idx = tid + kk * NTHR;     // 0..2047 : 16 rows x 128 cols
            const int r = idx >> 7, c = idx & 127;
            const int gi = (bBase + r) * TT + t;
            const float v = (c < VV) ? data_in[gi * VV + c]
                                     : mask_in[gi * VV + (c - VV)];
            Xlds[r * XS + c] = (__bf16)v;
        }
        if (tid < 16) {
            const int gi = (bBase + tid) * TT + t;
            const float t0 = times_in[gi];
            texist[tid] = t0;
            Xlds[tid * XS + 128] = (__bf16)((t + 1 < TT) ? times_in[gi + 1] - t0 : 0.0f);
        }
        __syncthreads();

        v8f accR  = (v8f)0.0f;
        v8f accZ  = (v8f)0.0f;
        v8f accNX = (v8f)0.0f;
        v8f accNH = (v8f)0.0f;

        // ---- x-GEMM: K=160 over Xlds, 1-deep pipelined pointer-bump loop ----
        {
            const __bf16* aP = aXb;
            const __bf16* rP = pXR;
            const __bf16* zP = pXZ;
            const __bf16* nP = pXN;
            v16bf a  = ldAf(aP);
            v16bf fR = ldBf(rP);
            v16bf fZ = ldBf(zP);
            v16bf fN = ldBf(nP);
#pragma unroll 2
            for (int kt = 0; kt < KTX - 1; ++kt) {
                aP += 32; rP += 512; zP += 512; nP += 512;
                v16bf a2  = ldAf(aP);
                v16bf fR2 = ldBf(rP);
                v16bf fZ2 = ldBf(zP);
                v16bf fN2 = ldBf(nP);
                accR  = WMMA_BF16(a, fR, accR);
                accZ  = WMMA_BF16(a, fZ, accZ);
                accNX = WMMA_BF16(a, fN, accNX);
                a = a2; fR = fR2; fZ = fZ2; fN = fN2;
            }
            accR  = WMMA_BF16(a, fR, accR);
            accZ  = WMMA_BF16(a, fZ, accZ);
            accNX = WMMA_BF16(a, fN, accNX);
        }
        // ---- h-GEMM: K=256 over Hbf, same structure ----
        {
            const __bf16* aP = aHb;
            const __bf16* rP = pHR;
            const __bf16* zP = pHZ;
            const __bf16* nP = pHN;
            v16bf a  = ldAf(aP);
            v16bf fR = ldBf(rP);
            v16bf fZ = ldBf(zP);
            v16bf fN = ldBf(nP);
#pragma unroll 2
            for (int kt = 0; kt < KTH - 1; ++kt) {
                aP += 32; rP += 512; zP += 512; nP += 512;
                v16bf a2  = ldAf(aP);
                v16bf fR2 = ldBf(rP);
                v16bf fZ2 = ldBf(zP);
                v16bf fN2 = ldBf(nP);
                accR  = WMMA_BF16(a, fR, accR);
                accZ  = WMMA_BF16(a, fZ, accZ);
                accNH = WMMA_BF16(a, fN, accNH);
                a = a2; fR = fR2; fZ = fZ2; fN = fN2;
            }
            accR  = WMMA_BF16(a, fR, accR);
            accZ  = WMMA_BF16(a, fZ, accZ);
            accNH = WMMA_BF16(a, fN, accNH);
        }
        __syncthreads();   // all waves done reading Hbf before rewriting it

        // ---- gates + masked state carry (each wave owns disjoint cols) ----
#pragma unroll
        for (int i = 0; i < 8; ++i) {
            float r = sigmoidf_(accR[i] + bR);
            float z = sigmoidf_(accZ[i] + bZ);
            float n = tanhf_(accNX[i] + bNX + r * (accNH[i] + bNH));
            float hold = hP[i * HS];
            float hnew = (1.0f - z) * n + z * hold;
            hnew = (teP[i] > 0.0f) ? hnew : hold;
            hP [i * HS] = hnew;
            hbP[i * HS] = (__bf16)hnew;
        }
        __syncthreads();   // h writes visible before next step's GEMM reads
    }

    // ---- write final h ----
    for (int idx = tid; idx < 16 * HH; idx += NTHR) {
        int r = idx >> 8, c = idx & 255;
        out[(bBase + r) * HH + c] = Hf[r * HS + c];
    }
}

extern "C" void kernel_launch(void* const* d_in, const int* in_sizes, int n_in,
                              void* d_out, int out_size, void* d_ws, size_t ws_size,
                              hipStream_t stream) {
    const float* times_in = (const float*)d_in[0];
    const float* data_in  = (const float*)d_in[1];
    const float* mask_in  = (const float*)d_in[2];
    const float* W_ih     = (const float*)d_in[3];
    const float* W_hh     = (const float*)d_in[4];
    const float* b_ih     = (const float*)d_in[5];
    const float* b_hh     = (const float*)d_in[6];
    float* out = (float*)d_out;

    __bf16* whhp = (__bf16*)d_ws;                                   // 393216 B
    __bf16* wihp = (__bf16*)((char*)d_ws + (size_t)NT3 * KTH * 32 * 16 * 2);

    pack_whh<<<(NT3 * KTH * 32 * 16 + 255) / 256, 256, 0, stream>>>(W_hh, whhp);
    pack_wih<<<(NT3 * KTX * 32 * 16 + 255) / 256, 256, 0, stream>>>(W_ih, wihp);
    grud_scan<<<BB / 16, NTHR, 0, stream>>>(times_in, data_in, mask_in,
                                            b_ih, b_hh, whhp, wihp, out);
}
